// EncoderBlock_87703232184339
// MI455X (gfx1250) — compile-verified, matrix operand order checked by reading, not run on hardware
//
#include <hip/hip_runtime.h>
#include <hip/hip_bf16.h>

// ---------------------------------------------------------------- constants
constexpr int B  = 8;
constexpr int S  = 1024;
constexpr int D  = 1024;
constexpr int H  = 16;
constexpr int DH = 64;
constexpr int FF = 4096;
constexpr int BS = B * S;       // 8192
constexpr int HD = H * DH;      // 1024

// ---------------------------------------------------------------- types
typedef __attribute__((ext_vector_type(16))) __bf16 bf16x16;
typedef __attribute__((ext_vector_type(8)))  __bf16 bf16x8;
typedef __attribute__((ext_vector_type(8)))  float  f32x8;

union BFrag { bf16x16 f; bf16x8 h[2]; };

// Load one 16x32 bf16 A/B fragment for v_wmma_f32_16x16x32_bf16.
// Caller passes p = base + (row + lane%16)*ld + k + (lane/16)*8.
// Lane holds K chunks {kb..kb+7} and {kb+16..kb+23} per the CDNA5 layout.
__device__ __forceinline__ bf16x16 load_frag16(const __hip_bfloat16* p) {
    BFrag u;
    u.h[0] = *reinterpret_cast<const bf16x8*>(p);
    u.h[1] = *reinterpret_cast<const bf16x8*>(p + 16);
    return u.f;
}

__device__ __forceinline__ f32x8 wmma_bf16(bf16x16 a, bf16x16 b, f32x8 c) {
    return __builtin_amdgcn_wmma_f32_16x16x32_bf16(
        /*neg_a=*/false, a, /*neg_b=*/false, b,
        /*c_mod=*/(short)0, c, /*reuse_a=*/false, /*reuse_b=*/false);
}

// ------------------------------------------------ async global->LDS support
#if defined(__has_builtin)
#  if __has_builtin(__builtin_amdgcn_global_load_async_to_lds_b128) && \
      __has_builtin(__builtin_amdgcn_s_wait_asynccnt)
#    define USE_ASYNC_LDS 1
#  endif
#endif
#ifndef USE_ASYNC_LDS
#  define USE_ASYNC_LDS 0
#endif

#if USE_ASYNC_LDS
#  define WAIT_ASYNC(n) __builtin_amdgcn_s_wait_asynccnt(n)
#else
#  define WAIT_ASYNC(n) ((void)0)
#endif

// Builtin parameter type: pointer to GCC-style 16-byte int vector,
// global (AS1) source / LDS (AS3) destination.
typedef int v4i128 __attribute__((vector_size(4 * sizeof(int))));
typedef __attribute__((address_space(1))) v4i128 gv4i128;
typedef __attribute__((address_space(3))) v4i128 lv4i128;

// Copy 16 bytes (8 bf16) global -> LDS.  Async (ASYNCcnt, bypasses VGPRs)
// when the gfx1250 builtin is available, else register-staged ds_store.
__device__ __forceinline__ void copy16_g2l(const __hip_bfloat16* g,
                                           __hip_bfloat16* l) {
#if USE_ASYNC_LDS
    __builtin_amdgcn_global_load_async_to_lds_b128(
        (gv4i128*)g, (lv4i128*)l, /*offset=*/0, /*cpol=*/0);
#else
    *reinterpret_cast<bf16x8*>(l) = *reinterpret_cast<const bf16x8*>(g);
#endif
}

// ---------------------------------------------------------------- converters
__global__ void k_cvt_bf16(const float* __restrict__ in,
                           __hip_bfloat16* __restrict__ out, int n) {
    int i = blockIdx.x * blockDim.x + threadIdx.x;
    int stride = gridDim.x * blockDim.x;
    for (; i < n; i += stride) out[i] = __float2bfloat16(in[i]);
}

// w: [H, D, DH] f32  ->  out: [H*DH, D] bf16  (row n = h*DH+e holds w[h,:,e])
__global__ void k_repack_qkv(const float* __restrict__ w,
                             __hip_bfloat16* __restrict__ out) {
    int i = blockIdx.x * blockDim.x + threadIdx.x;
    int stride = gridDim.x * blockDim.x;
    const int total = H * D * DH;
    for (; i < total; i += stride) {
        int e = i % DH;
        int d = (i / DH) % D;
        int h = i / (D * DH);
        out[(size_t)(h * DH + e) * D + d] = __float2bfloat16(w[i]);
    }
}

// in: [R, C] f32 -> out: [C, R] bf16
__global__ void k_transpose_cvt(const float* __restrict__ in,
                                __hip_bfloat16* __restrict__ out, int R, int C) {
    int i = blockIdx.x * blockDim.x + threadIdx.x;
    int stride = gridDim.x * blockDim.x;
    const int total = R * C;
    for (; i < total; i += stride) {
        int r = i / C, c = i % C;
        out[(size_t)c * R + r] = __float2bfloat16(in[i]);
    }
}

// ---------------------------------------------------------------- GEMM
// C[M,N] = act(A[M,K] @ Bt[N,K]^T + bias + resid)
// 8 waves/block; block tile 128x64; wave tile 32x32 (2x2 WMMA).
// K staged through LDS in 64-wide double-buffered stages fed by async
// global->LDS copies (ASYNCcnt); fragments read back with ds_load_b128.
// mode 0: f32 row-major   mode 1: bf16 [b,h,s,e]   mode 2: bf16 [b,h,e,s]
// mode 3: bf16 row-major
__global__ __launch_bounds__(256)
void k_gemm(const __hip_bfloat16* __restrict__ A,
            const __hip_bfloat16* __restrict__ Bt,
            const float* __restrict__ bias,
            const float* __restrict__ resid,
            float* __restrict__ outF,
            __hip_bfloat16* __restrict__ outB,
            int M, int N, int K, int mode, int relu) {
    const int lane = threadIdx.x & 31;
    const int wave = threadIdx.x >> 5;
    const int lr = lane & 15, lh = lane >> 4;
    const int wm = wave >> 1;           // 0..3  (M direction)
    const int wn = wave & 1;            // 0..1  (N direction)
    const int mBase = blockIdx.x * 128;
    const int nBase = blockIdx.y * 64;
    const int m0 = mBase + wm * 32;
    const int n0 = nBase + wn * 32;

    // double-buffered staging: A 128x64, B 64x64 per stage (24 KB/stage)
    __shared__ __align__(16) __hip_bfloat16 sA[2][128 * 64];
    __shared__ __align__(16) __hip_bfloat16 sB[2][64 * 64];

    // 1536 x 16B chunks per stage, 6 per thread
    auto issue_stage = [&](int s) {
        const int k0 = s * 64;
        const int buf = s & 1;
        #pragma unroll
        for (int i = 0; i < 6; ++i) {
            int c = i * 256 + (int)threadIdx.x;
            if (c < 1024) {                       // A tile: 128 rows x 64
                int row = c >> 3, off = (c & 7) * 8;
                copy16_g2l(A + (size_t)(mBase + row) * K + k0 + off,
                           &sA[buf][row * 64 + off]);
            } else {                              // B tile: 64 rows x 64
                int cb = c - 1024;
                int row = cb >> 3, off = (cb & 7) * 8;
                copy16_g2l(Bt + (size_t)(nBase + row) * K + k0 + off,
                           &sB[buf][row * 64 + off]);
            }
        }
    };

    f32x8 acc[2][2] = {};
    const int ksteps = K >> 6;          // K is a multiple of 64 here

    issue_stage(0);
    for (int s = 0; s < ksteps; ++s) {
        if (s + 1 < ksteps) {
            issue_stage(s + 1);
            WAIT_ASYNC(6);              // retire stage s (this wave)
        } else {
            WAIT_ASYNC(0);
        }
        __syncthreads();                // stage s visible from all waves

        const __hip_bfloat16* pa = &sA[s & 1][(wm * 32 + lr) * 64 + lh * 8];
        const __hip_bfloat16* pb = &sB[s & 1][(wn * 32 + lr) * 64 + lh * 8];
        #pragma unroll
        for (int ks = 0; ks < 64; ks += 32) {
            bf16x16 a0 = load_frag16(pa + ks);
            bf16x16 a1 = load_frag16(pa + 16 * 64 + ks);
            bf16x16 b0 = load_frag16(pb + ks);
            bf16x16 b1 = load_frag16(pb + 16 * 64 + ks);
            acc[0][0] = wmma_bf16(a0, b0, acc[0][0]);
            acc[0][1] = wmma_bf16(a0, b1, acc[0][1]);
            acc[1][0] = wmma_bf16(a1, b0, acc[1][0]);
            acc[1][1] = wmma_bf16(a1, b1, acc[1][1]);
        }
        __syncthreads();                // done reading before s+2 overwrites
    }

    for (int i = 0; i < 2; i++)
        for (int j = 0; j < 2; j++)
            for (int r = 0; r < 8; r++) {
                int m = m0 + i * 16 + r + lh * 8;   // C layout: M = r + 8*half
                int n = n0 + j * 16 + lr;           //           N = lane%16
                float v = acc[i][j][r];
                if (bias)  v += bias[n];
                if (resid) v += resid[(size_t)m * N + n];
                if (relu)  v = fmaxf(v, 0.0f);
                if (mode == 0) {
                    outF[(size_t)m * N + n] = v;
                } else if (mode == 3) {
                    outB[(size_t)m * N + n] = __float2bfloat16(v);
                } else {
                    int b = m >> 10, sidx = m & 1023;  // S = 1024
                    int h = n >> 6,  e = n & 63;       // DH = 64
                    size_t idx = (mode == 1)
                        ? ((size_t)((b * H + h) * S + sidx) * DH + e)
                        : ((size_t)((b * H + h) * DH + e) * S + sidx);
                    outB[idx] = __float2bfloat16(v);
                }
            }
}

// ---------------------------------------------------------------- attention
// Flash-attention: each wave owns 16 query rows of one (b,h); streams 32 keys
// per iteration, online softmax, ctx accumulated via WMMA against Vt [DH,S].
__global__ __launch_bounds__(128)
void k_attn(const __hip_bfloat16* __restrict__ Q,
            const __hip_bfloat16* __restrict__ Kc,
            const __hip_bfloat16* __restrict__ Vt,
            __hip_bfloat16* __restrict__ ctx) {
    const int lane = threadIdx.x & 31;
    const int wave = threadIdx.x >> 5;
    const int lr = lane & 15, lh = lane >> 4;
    const int gw = blockIdx.x * 4 + wave;       // 8192 waves total
    const int bh = gw >> 6;                     // (b*H + h)
    const int m0 = (gw & 63) << 4;              // query tile base

    const __hip_bfloat16* q  = Q  + (size_t)bh * S * DH;
    const __hip_bfloat16* kc = Kc + (size_t)bh * S * DH;
    const __hip_bfloat16* vt = Vt + (size_t)bh * DH * S;

    // Q fragments for k-steps 0 and 32 of DH=64
    const __hip_bfloat16* qp = q + (size_t)(m0 + lr) * DH + lh * 8;
    bf16x16 qa0 = load_frag16(qp);
    bf16x16 qa1 = load_frag16(qp + 32);

    f32x8 acc[4] = {};
    float rmax[8], rsum[8];
    for (int r = 0; r < 8; r++) { rmax[r] = -3.0e38f; rsum[r] = 0.0f; }

    // per-wave LDS scratch: P tile stashed as plain [16 rows][32 cols] bf16,
    // reloaded in A-fragment layout for the P@V WMMA.
    __shared__ __align__(16) __hip_bfloat16 pbuf[4][16 * 32];
    __hip_bfloat16* pw = pbuf[wave];

    for (int j0 = 0; j0 < S; j0 += 32) {
        f32x8 sc[2];
        for (int t = 0; t < 2; t++) {
            const __hip_bfloat16* kp =
                kc + (size_t)(j0 + t * 16 + lr) * DH + lh * 8;
            bf16x16 kb0 = load_frag16(kp);
            bf16x16 kb1 = load_frag16(kp + 32);
            f32x8 z = {};
            z = wmma_bf16(qa0, kb0, z);
            z = wmma_bf16(qa1, kb1, z);
            sc[t] = z * 0.125f;                 // 1/sqrt(DH)
        }

        // tile row-max: reduce over the 16 lanes of each half (xor<=8 stays
        // within a half, matching C-layout rows r / r+8).
        float tm[8];
        for (int r = 0; r < 8; r++) tm[r] = fmaxf(sc[0][r], sc[1][r]);
        for (int off = 1; off < 16; off <<= 1)
            for (int r = 0; r < 8; r++)
                tm[r] = fmaxf(tm[r], __shfl_xor(tm[r], off, 32));

        for (int r = 0; r < 8; r++) {
            float nm = fmaxf(rmax[r], tm[r]);
            float scale = __expf(rmax[r] - nm);
            rmax[r] = nm;
            rsum[r] *= scale;
            for (int n = 0; n < 4; n++) acc[n][r] *= scale;
            float p0 = __expf(sc[0][r] - nm);
            float p1 = __expf(sc[1][r] - nm);
            rsum[r] += p0 + p1;                 // per-lane partials; reduce once
            sc[0][r] = p0;
            sc[1][r] = p1;
        }

        // stash P (C layout) -> LDS plain matrix
        for (int t = 0; t < 2; t++)
            for (int r = 0; r < 8; r++)
                pw[(r + lh * 8) * 32 + t * 16 + lr] =
                    __float2bfloat16(sc[t][r]);
        __syncthreads();

        // reload P as A-fragment (lane row = lr, K chunks lh*8 / lh*8+16)
        bf16x16 pa = load_frag16(pw + lr * 32 + lh * 8);
        for (int n = 0; n < 4; n++) {
            bf16x16 vb =
                load_frag16(vt + (size_t)(n * 16 + lr) * S + j0 + lh * 8);
            acc[n] = wmma_bf16(pa, vb, acc[n]);
        }
        __syncthreads();
    }

    for (int off = 1; off < 16; off <<= 1)
        for (int r = 0; r < 8; r++)
            rsum[r] += __shfl_xor(rsum[r], off, 32);

    const int b = bh >> 4, h = bh & 15;
    for (int r = 0; r < 8; r++) {
        float inv = 1.0f / rsum[r];
        int si = m0 + r + lh * 8;
        for (int n = 0; n < 4; n++) {
            int e = n * 16 + lr;
            ctx[(size_t)(b * S + si) * HD + h * DH + e] =
                __float2bfloat16(acc[n][r] * inv);
        }
    }
}

// ---------------------------------------------------------------- layernorm
__global__ __launch_bounds__(256)
void k_layernorm(const float* __restrict__ in,
                 const float* __restrict__ g,
                 const float* __restrict__ be,
                 float* __restrict__ outF,
                 __hip_bfloat16* __restrict__ outB) {
    const int row = blockIdx.x;
    const float* p = in + (size_t)row * D;
    float s = 0.0f, ss = 0.0f;
    for (int i = threadIdx.x; i < D; i += 256) {
        float v = p[i];
        s += v;
        ss += v * v;
    }
    for (int off = 16; off > 0; off >>= 1) {
        s  += __shfl_xor(s,  off, 32);
        ss += __shfl_xor(ss, off, 32);
    }
    __shared__ float red[2][8];
    const int wave = threadIdx.x >> 5, lane = threadIdx.x & 31;
    if (lane == 0) { red[0][wave] = s; red[1][wave] = ss; }
    __syncthreads();
    if (threadIdx.x == 0) {
        float ts = 0.0f, tss = 0.0f;
        for (int w = 0; w < 8; w++) { ts += red[0][w]; tss += red[1][w]; }
        red[0][0] = ts;
        red[1][0] = tss;
    }
    __syncthreads();
    float mu  = red[0][0] * (1.0f / D);
    float var = red[1][0] * (1.0f / D) - mu * mu;
    float inv = rsqrtf(var + 1e-5f);
    for (int i = threadIdx.x; i < D; i += 256) {
        float y = (p[i] - mu) * inv * g[i] + be[i];
        outF[(size_t)row * D + i] = y;
        if (outB) outB[(size_t)row * D + i] = __float2bfloat16(y);
    }
}

// ---------------------------------------------------------------- launcher
extern "C" void kernel_launch(void* const* d_in, const int* in_sizes, int n_in,
                              void* d_out, int out_size, void* d_ws,
                              size_t ws_size, hipStream_t stream) {
    const float* x   = (const float*)d_in[0];
    const float* wq  = (const float*)d_in[1];
    const float* bq  = (const float*)d_in[2];
    const float* wk  = (const float*)d_in[3];
    const float* bk  = (const float*)d_in[4];
    const float* wv  = (const float*)d_in[5];
    const float* bv  = (const float*)d_in[6];
    const float* wo  = (const float*)d_in[7];
    const float* bo  = (const float*)d_in[8];
    const float* w1  = (const float*)d_in[9];
    const float* b1  = (const float*)d_in[10];
    const float* w2  = (const float*)d_in[11];
    const float* b2  = (const float*)d_in[12];
    const float* g1  = (const float*)d_in[13];
    const float* be1 = (const float*)d_in[14];
    const float* g2  = (const float*)d_in[15];
    const float* be2 = (const float*)d_in[16];
    float* out = (float*)d_out;

    char* ws = (char*)d_ws;
    size_t off = 0;
    auto alloc = [&](size_t bytes) {
        char* p = ws + off;
        off += (bytes + 255) & ~(size_t)255;
        return p;
    };
    __hip_bfloat16* xb    = (__hip_bfloat16*)alloc((size_t)BS * D * 2);
    __hip_bfloat16* wq_t  = (__hip_bfloat16*)alloc((size_t)HD * D * 2);
    __hip_bfloat16* wk_t  = (__hip_bfloat16*)alloc((size_t)HD * D * 2);
    __hip_bfloat16* wv_t  = (__hip_bfloat16*)alloc((size_t)HD * D * 2);
    __hip_bfloat16* wo_t  = (__hip_bfloat16*)alloc((size_t)D * HD * 2);
    __hip_bfloat16* w1_t  = (__hip_bfloat16*)alloc((size_t)FF * D * 2);
    __hip_bfloat16* w2_t  = (__hip_bfloat16*)alloc((size_t)D * FF * 2);
    __hip_bfloat16* q_b   = (__hip_bfloat16*)alloc((size_t)B * H * S * DH * 2);
    __hip_bfloat16* k_b   = (__hip_bfloat16*)alloc((size_t)B * H * S * DH * 2);
    __hip_bfloat16* vt_b  = (__hip_bfloat16*)alloc((size_t)B * H * DH * S * 2);
    __hip_bfloat16* ctx_b = (__hip_bfloat16*)alloc((size_t)BS * HD * 2);
    float*          resbf = (float*)alloc((size_t)BS * D * 4);
    float*          y1f   = (float*)alloc((size_t)BS * D * 4);
    __hip_bfloat16* y1b   = (__hip_bfloat16*)alloc((size_t)BS * D * 2);
    __hip_bfloat16* ffn1  = (__hip_bfloat16*)alloc((size_t)BS * FF * 2);
    (void)ws_size; (void)in_sizes; (void)n_in; (void)out_size;

    // ---- precision conversion / weight re-layout (bf16, transposed) ----
    k_cvt_bf16<<<1024, 256, 0, stream>>>(x, xb, BS * D);
    k_repack_qkv<<<1024, 256, 0, stream>>>(wq, wq_t);
    k_repack_qkv<<<1024, 256, 0, stream>>>(wk, wk_t);
    k_repack_qkv<<<1024, 256, 0, stream>>>(wv, wv_t);
    k_transpose_cvt<<<1024, 256, 0, stream>>>(wo, wo_t, HD, D);
    k_transpose_cvt<<<2048, 256, 0, stream>>>(w1, w1_t, D, FF);
    k_transpose_cvt<<<2048, 256, 0, stream>>>(w2, w2_t, FF, D);

    // ---- QKV projections (WMMA) ----
    dim3 blk(256);
    k_gemm<<<dim3(BS / 128, HD / 64), blk, 0, stream>>>(
        xb, wq_t, bq, nullptr, nullptr, q_b, BS, HD, D, /*mode=*/1, 0);
    k_gemm<<<dim3(BS / 128, HD / 64), blk, 0, stream>>>(
        xb, wk_t, bk, nullptr, nullptr, k_b, BS, HD, D, /*mode=*/1, 0);
    k_gemm<<<dim3(BS / 128, HD / 64), blk, 0, stream>>>(
        xb, wv_t, bv, nullptr, nullptr, vt_b, BS, HD, D, /*mode=*/2, 0);

    // ---- flash attention (WMMA) ----
    k_attn<<<(B * H * (S / 16)) / 4, 128, 0, stream>>>(q_b, k_b, vt_b, ctx_b);

    // ---- output projection + residual, LN1 ----
    k_gemm<<<dim3(BS / 128, D / 64), blk, 0, stream>>>(
        ctx_b, wo_t, bo, x, resbf, nullptr, BS, D, HD, /*mode=*/0, 0);
    k_layernorm<<<BS, 256, 0, stream>>>(resbf, g1, be1, y1f, y1b);

    // ---- FFN (WMMA) + residual, LN2 ----
    k_gemm<<<dim3(BS / 128, FF / 64), blk, 0, stream>>>(
        y1b, w1_t, b1, nullptr, nullptr, ffn1, BS, FF, D, /*mode=*/3, /*relu=*/1);
    k_gemm<<<dim3(BS / 128, D / 64), blk, 0, stream>>>(
        ffn1, w2_t, b2, y1f, resbf, nullptr, BS, D, FF, /*mode=*/0, 0);
    k_layernorm<<<BS, 256, 0, stream>>>(resbf, g2, be2, out, nullptr);
}
